// PointNet2Encoder_24567212933991
// MI455X (gfx1250) — matrix operand order repaired and checked by reading, not compiled, not run on hardware
//
#include <hip/hip_runtime.h>
#include <hip/hip_bf16.h>
#include <math.h>

typedef __attribute__((ext_vector_type(16))) _Float16 v16h;
typedef __attribute__((ext_vector_type(8)))  _Float16 v8h;
typedef __attribute__((ext_vector_type(8)))  float    v8f;

#define DEV __device__ __forceinline__

static constexpr float kEps = 1e-5f;
static constexpr int kB  = 16;
static constexpr int kG1 = 512;   // FPS-1 samples
static constexpr int kG2 = 128;   // FPS-2 samples
static constexpr int kS2 = 64;    // group size stage 2

// ---------------------------------------------------------------------------
// BN(eval) folding:  y = (x@W + b - mean)*rsqrt(var+eps)*gamma + beta
//                      = acc*scale + shift
// ---------------------------------------------------------------------------
__global__ void bn_fold_kernel(const float* __restrict__ bias,
                               const float* __restrict__ gamma,
                               const float* __restrict__ beta,
                               const float* __restrict__ mean,
                               const float* __restrict__ var,
                               float* __restrict__ scale,
                               float* __restrict__ shift, int n) {
  int i = blockIdx.x * blockDim.x + threadIdx.x;
  if (i < n) {
    float s = gamma[i] * rsqrtf(var[i] + kEps);
    scale[i] = s;
    shift[i] = (bias[i] - mean[i]) * s + beta[i];
  }
}

__global__ void init_out_kernel(float* __restrict__ out, int n) {
  int i = blockIdx.x * blockDim.x + threadIdx.x;
  if (i < n) out[i] = 0.0f;
}

// ---------------------------------------------------------------------------
// Weight pre-pack: f32 row-major (cout x cin) -> f16 WMMA A-fragment-linear.
// One thread per (fragment, lane).  Fragment (ct,kk) covers rows ct*16..+15,
// K = kk*32..+31.  A k order packed here (contiguous; B loads use the same):
//   lane lr (row), half hh: elems[0..7] = K kb..kb+7, elems[8..15] = kb+8..
//   with kb = kk*32 + 16*hh.  Out-of-range cin zero-filled here, once.
// In the GEMM kernels an A fragment is then ONE aligned 32B load per lane.
// ---------------------------------------------------------------------------
__global__ void wfrag_kernel(const float* __restrict__ W, int cout, int cin,
                             int nkk, _Float16* __restrict__ out) {
  int idx = blockIdx.x * blockDim.x + threadIdx.x;
  int nelem = (cout >> 4) * nkk * 32;
  if (idx >= nelem) return;
  int frag = idx >> 5, lane = idx & 31;
  int ct = frag / nkk, kk = frag - ct * nkk;
  int lr = lane & 15, hh = lane >> 4;
  const float* row = W + (size_t)(ct * 16 + lr) * cin;
  const int kb = kk * 32 + hh * 16;
  v16h a;
#pragma unroll
  for (int i = 0; i < 8; ++i) {
    int ka = kb + i, kc = kb + 8 + i;
    a[i]     = (ka < cin) ? (_Float16)row[ka] : (_Float16)0.0f;
    a[i + 8] = (kc < cin) ? (_Float16)row[kc] : (_Float16)0.0f;
  }
  *(v16h*)(out + (size_t)idx * 16) = a;
}

// ---------------------------------------------------------------------------
// Farthest point sampling: one workgroup per batch, LDS running-distance,
// tree argmax with lowest-index tie-break (matches jnp.argmax semantics).
// ---------------------------------------------------------------------------
template <int N, int NT>
__global__ void __launch_bounds__(NT)
fps_kernel(const float* __restrict__ pts, float* __restrict__ centers, int nSamples) {
  __shared__ float dist[N];
  __shared__ float rv[NT];
  __shared__ int   ri[NT];
  __shared__ float lastP[3];
  const int bb  = blockIdx.x;
  const int tid = threadIdx.x;
  const float* P = pts + (size_t)bb * N * 3;
  float* C = centers + (size_t)bb * nSamples * 3;

  for (int j = tid; j < N; j += NT) dist[j] = 3.4e38f;
  if (tid == 0) {
    lastP[0] = P[0]; lastP[1] = P[1]; lastP[2] = P[2];
    C[0] = P[0]; C[1] = P[1]; C[2] = P[2];
  }
  __syncthreads();

  for (int it = 1; it < nSamples; ++it) {
    const float lx = lastP[0], ly = lastP[1], lz = lastP[2];
    float best = -1.0f; int bi = N;
    for (int j = tid; j < N; j += NT) {
      float dx = P[j*3+0] - lx, dy = P[j*3+1] - ly, dz = P[j*3+2] - lz;
      float d  = dx*dx + dy*dy + dz*dz;
      float dm = fminf(dist[j], d);
      dist[j] = dm;
      if (dm > best) { best = dm; bi = j; }   // strict '>' keeps lowest index
    }
    rv[tid] = best; ri[tid] = bi;
    __syncthreads();
    for (int s = NT / 2; s > 0; s >>= 1) {
      if (tid < s) {
        float ov = rv[tid + s]; int oi = ri[tid + s];
        if (ov > rv[tid] || (ov == rv[tid] && oi < ri[tid])) { rv[tid] = ov; ri[tid] = oi; }
      }
      __syncthreads();
    }
    if (tid == 0) {
      int sel = ri[0];
      float x = P[sel*3+0], y = P[sel*3+1], z = P[sel*3+2];
      lastP[0] = x; lastP[1] = y; lastP[2] = z;
      C[it*3+0] = x; C[it*3+1] = y; C[it*3+2] = z;
    }
    __syncthreads();
  }
}

// ---------------------------------------------------------------------------
// kNN (64 nearest of 512) + center-relative grouping, output f16 compact.
// Iterative argmin with lowest-index tie-break == lax.top_k(-d) selection set.
// ---------------------------------------------------------------------------
__global__ void __launch_bounds__(64)
knn_group_kernel(const float* __restrict__ c1, const float* __restrict__ c2,
                 _Float16* __restrict__ nb2h) {
  __shared__ float dd[kG1];
  __shared__ float rv[64];
  __shared__ int   ri[64];
  const int gid = blockIdx.x;        // b*128 + g
  const int bb  = gid >> 7;
  const int tid = threadIdx.x;
  const float* P = c1 + (size_t)bb * kG1 * 3;
  const float cx = c2[gid*3+0], cy = c2[gid*3+1], cz = c2[gid*3+2];

  for (int j = tid; j < kG1; j += 64) {
    float dx = P[j*3+0] - cx, dy = P[j*3+1] - cy, dz = P[j*3+2] - cz;
    dd[j] = dx*dx + dy*dy + dz*dz;
  }
  __syncthreads();

  _Float16* O = nb2h + (size_t)gid * (kS2 * 3);   // [s][3]
  for (int k = 0; k < kS2; ++k) {
    float best = 3.4e38f; int bi = kG1;
    for (int j = tid; j < kG1; j += 64) {
      float v = dd[j];
      if (v < best) { best = v; bi = j; }
    }
    rv[tid] = best; ri[tid] = bi;
    __syncthreads();
    for (int s = 32; s > 0; s >>= 1) {
      if (tid < s) {
        float ov = rv[tid + s]; int oi = ri[tid + s];
        if (ov < rv[tid] || (ov == rv[tid] && oi < ri[tid])) { rv[tid] = ov; ri[tid] = oi; }
      }
      __syncthreads();
    }
    if (tid == 0) {
      int sel = ri[0];
      dd[sel] = 3.4e38f;
      O[k*3+0] = (_Float16)(P[sel*3+0] - cx);
      O[k*3+1] = (_Float16)(P[sel*3+1] - cy);
      O[k*3+2] = (_Float16)(P[sel*3+2] - cz);
    }
    __syncthreads();
  }
}

// ---------------------------------------------------------------------------
// WMMA fragment helpers (V_WMMA_F32_16X16X32_F16, wave32).
// A: pre-packed fragment-linear f16 -> one 32B global load per lane.
// B: activations f16 point-major   -> one 32B LDS load per lane.
// C/D f32: VGPR v <-> row v + 8*hh, lane lr <-> column.
// ---------------------------------------------------------------------------
DEV v16h load_a_pre(const _Float16* __restrict__ A, int ct, int nkk, int kk, int lane) {
  return *(const v16h*)(A + ((size_t)(ct * nkk + kk) * 32 + lane) * 16);
}

DEV v16h load_b_act(const _Float16* X, int cstride, int p0, int k0, int lr, int hh) {
  return *(const v16h*)(X + (size_t)(p0 + lr) * cstride + k0 + hh * 16);
}

DEV v8f wmma_f16(v16h a, v16h b, v8f c) {
  return __builtin_amdgcn_wmma_f32_16x16x32_f16(false, a, false, b, (short)0, c,
                                                false, false);
}

DEV v8f bn_relu(v8f acc, const float* __restrict__ sc, const float* __restrict__ sh,
                int r0, int hh) {
  v8f y;
#pragma unroll
  for (int v = 0; v < 8; ++v) {
    int c = r0 + hh * 8 + v;
    y[v] = fmaxf(acc[v] * sc[c] + sh[c], 0.0f);
  }
  return y;
}

DEV void store_d_act(_Float16* Y, int cstride, int p0, int r0, int lr, int hh, v8f y) {
  v8h h;
#pragma unroll
  for (int v = 0; v < 8; ++v) h[v] = (_Float16)y[v];
  *(v8h*)(Y + (size_t)(p0 + lr) * cstride + r0 + hh * 8) = h;  // 16B aligned
}

// ---------------------------------------------------------------------------
// MLP2 [3->128->128->256] fused per (b,g) group of 64 points, max over group.
// Activations LDS-resident f16 point-major; output f2 as f16 point-major.
// ---------------------------------------------------------------------------
__global__ void __launch_bounds__(256)
mlp2_kernel(const _Float16* __restrict__ nb2h,
            const _Float16* __restrict__ A1, const _Float16* __restrict__ A2,
            const _Float16* __restrict__ A3,
            const float* __restrict__ sc1, const float* __restrict__ sh1,
            const float* __restrict__ sc2, const float* __restrict__ sh2,
            const float* __restrict__ sc3, const float* __restrict__ sh3,
            _Float16* __restrict__ act3) {
  __shared__ _Float16 X0[kS2 * 32];    // 4 KB, channels 3..31 zero-padded
  __shared__ _Float16 Y1[kS2 * 128];   // 16 KB
  __shared__ _Float16 Y2[kS2 * 128];   // 16 KB
  const int gid = blockIdx.x;          // b*128 + g
  const int tid = threadIdx.x;
  const int w = tid >> 5, lane = tid & 31, lr = lane & 15, hh = lane >> 4;

  __builtin_prefetch(A2 + (size_t)lane * 64, 0, 0);
  __builtin_prefetch(A3 + (size_t)lane * 64, 0, 0);

  for (int i = tid; i < kS2 * 32; i += 256) X0[i] = (_Float16)0.0f;
  __syncthreads();
  const _Float16* src = nb2h + (size_t)gid * (kS2 * 3);
  for (int i = tid; i < kS2 * 3; i += 256) {
    int s = i / 3, c = i - s * 3;
    X0[s * 32 + c] = src[i];
  }
  __syncthreads();

  // layer 1: 3 (zero-padded to 32) -> 128
  for (int t = 0; t < 4; ++t) {
    int tk = w * 4 + t, ct = tk >> 2, pt = tk & 3;   // 8 ct x 4 pt tiles
    v8f acc = {};
    acc = wmma_f16(load_a_pre(A1, ct, 1, 0, lane),
                   load_b_act(X0, 32, pt * 16, 0, lr, hh), acc);
    store_d_act(Y1, 128, pt * 16, ct * 16, lr, hh, bn_relu(acc, sc1, sh1, ct * 16, hh));
  }
  __syncthreads();

  // layer 2: 128 -> 128
  for (int t = 0; t < 4; ++t) {
    int tk = w * 4 + t, ct = tk >> 2, pt = tk & 3;
    v8f acc = {};
#pragma unroll
    for (int kk = 0; kk < 4; ++kk)
      acc = wmma_f16(load_a_pre(A2, ct, 4, kk, lane),
                     load_b_act(Y1, 128, pt * 16, kk * 32, lr, hh), acc);
    store_d_act(Y2, 128, pt * 16, ct * 16, lr, hh, bn_relu(acc, sc2, sh2, ct * 16, hh));
  }
  __syncthreads();

  // layer 3: 128 -> 256, fused max over the 64 group points
  for (int j = 0; j < 2; ++j) {
    int ct = w + j * 8;                 // 16 cout tiles
    v8f macc = {};                      // post-ReLU values are >= 0
    for (int pt = 0; pt < 4; ++pt) {
      v8f acc = {};
#pragma unroll
      for (int kk = 0; kk < 4; ++kk)
        acc = wmma_f16(load_a_pre(A3, ct, 4, kk, lane),
                       load_b_act(Y2, 128, pt * 16, kk * 32, lr, hh), acc);
      v8f y = bn_relu(acc, sc3, sh3, ct * 16, hh);
#pragma unroll
      for (int v = 0; v < 8; ++v) macc[v] = fmaxf(macc[v], y[v]);
    }
#pragma unroll
    for (int off = 1; off < 16; off <<= 1) {
#pragma unroll
      for (int v = 0; v < 8; ++v) macc[v] = fmaxf(macc[v], __shfl_xor(macc[v], off));
    }
    if (lr == 0) {
      v8h h;
#pragma unroll
      for (int v = 0; v < 8; ++v) h[v] = (_Float16)macc[v];
      *(v8h*)(act3 + (size_t)gid * 256 + ct * 16 + hh * 8) = h;
    }
  }
}

// ---------------------------------------------------------------------------
// MLP3 [256->256->512->1024] per 16-point chunk, global max via atomics on
// float bits (values are post-ReLU >= 0, d_out initialized to 0).
// ---------------------------------------------------------------------------
__global__ void __launch_bounds__(256)
mlp3_kernel(const _Float16* __restrict__ act3,
            const _Float16* __restrict__ A4, const _Float16* __restrict__ A5,
            const _Float16* __restrict__ A6,
            const float* __restrict__ sc4, const float* __restrict__ sh4,
            const float* __restrict__ sc5, const float* __restrict__ sh5,
            const float* __restrict__ sc6, const float* __restrict__ sh6,
            float* __restrict__ out) {
  __shared__ _Float16 Xc[16 * 256];   // 8 KB
  __shared__ _Float16 Y1[16 * 256];   // 8 KB
  __shared__ _Float16 Y2[16 * 512];   // 16 KB
  const int blk = blockIdx.x;         // b*8 + chunk  (global point tile)
  const int bb  = blk >> 3;
  const int tid = threadIdx.x;
  const int w = tid >> 5, lane = tid & 31, lr = lane & 15, hh = lane >> 4;

  __builtin_prefetch(A6 + (size_t)lane * 64, 0, 0);

  const _Float16* src = act3 + (size_t)blk * 16 * 256;
  for (int i = tid; i < 16 * 256; i += 256) Xc[i] = src[i];
  __syncthreads();

  // layer 1: 256 -> 256
  for (int j = 0; j < 2; ++j) {
    int ct = w + j * 8;
    v8f acc = {};
#pragma unroll
    for (int kk = 0; kk < 8; ++kk)
      acc = wmma_f16(load_a_pre(A4, ct, 8, kk, lane),
                     load_b_act(Xc, 256, 0, kk * 32, lr, hh), acc);
    store_d_act(Y1, 256, 0, ct * 16, lr, hh, bn_relu(acc, sc4, sh4, ct * 16, hh));
  }
  __syncthreads();

  // layer 2: 256 -> 512
  for (int j = 0; j < 4; ++j) {
    int ct = w + j * 8;
    v8f acc = {};
#pragma unroll
    for (int kk = 0; kk < 8; ++kk)
      acc = wmma_f16(load_a_pre(A5, ct, 8, kk, lane),
                     load_b_act(Y1, 256, 0, kk * 32, lr, hh), acc);
    store_d_act(Y2, 512, 0, ct * 16, lr, hh, bn_relu(acc, sc5, sh5, ct * 16, hh));
  }
  __syncthreads();

  // layer 3: 512 -> 1024 + batch-wide max
  for (int j = 0; j < 8; ++j) {
    int ct = w + j * 8;
    v8f acc = {};
#pragma unroll
    for (int kk = 0; kk < 16; ++kk)
      acc = wmma_f16(load_a_pre(A6, ct, 16, kk, lane),
                     load_b_act(Y2, 512, 0, kk * 32, lr, hh), acc);
    v8f y = bn_relu(acc, sc6, sh6, ct * 16, hh);
#pragma unroll
    for (int off = 1; off < 16; off <<= 1) {
#pragma unroll
      for (int v = 0; v < 8; ++v) y[v] = fmaxf(y[v], __shfl_xor(y[v], off));
    }
    if (lr == 0) {
#pragma unroll
      for (int v = 0; v < 8; ++v)
        atomicMax((int*)(out + (size_t)bb * 1024 + ct * 16 + hh * 8 + v),
                  __float_as_int(y[v]));
    }
  }
}

// ---------------------------------------------------------------------------
// Host launch. Input order (setup_inputs dict order): x, params1 (3 layers x
// {w,b,gamma,beta,mean,var}), params2, params3.  params1 is dead code in the
// reference (_f1 unused) and is skipped entirely.
// ---------------------------------------------------------------------------
extern "C" void kernel_launch(void* const* d_in, const int* in_sizes, int n_in,
                              void* d_out, int out_size, void* d_ws, size_t ws_size,
                              hipStream_t stream) {
  (void)in_sizes; (void)n_in; (void)ws_size;
  const float* x = (const float*)d_in[0];
  auto L = [&](int base, int j) { return (const float*)d_in[base + j]; };
  const int p2 = 19, p3 = 37;   // first leaf index of params2 / params3

  char* ws = (char*)d_ws;
  float*    c1    = (float*)(ws + 0);            // 16*512*3 f32
  float*    c2    = (float*)(ws + 98304);        // 16*128*3 f32
  _Float16* nb2h  = (_Float16*)(ws + 122880);    // 2048*64*3 f16
  _Float16* act3  = (_Float16*)(ws + 909312);    // 2048*256  f16 (f2, point-major)
  float*    bnbuf = (float*)(ws + 1957888);      // folded BN scale/shift
  char*     wbase = ws + 1976320;                // pre-packed weight fragments

  // layer geometry: cout, cin, nkk (=ceil(cin/32)), weight leaf base
  const int cout_[6] = {128, 128, 256, 256, 512, 1024};
  const int cin_[6]  = {3,   128, 128, 256, 256, 512};
  const int nkk_[6]  = {1,   4,   4,   8,   8,   16};
  const int bases[6] = {p2, p2 + 6, p2 + 12, p3, p3 + 6, p3 + 12};

  float* sc[6]; float* sh[6];
  int off = 0;
  for (int i = 0; i < 6; ++i) { sc[i] = bnbuf + off; off += cout_[i];
                                sh[i] = bnbuf + off; off += cout_[i]; }

  _Float16* A[6];
  size_t woff = 0;
  for (int i = 0; i < 6; ++i) {
    A[i] = (_Float16*)(wbase + woff);
    woff += (size_t)(cout_[i] / 16) * nkk_[i] * 32 * 16 * sizeof(_Float16);
  }

  for (int i = 0; i < 6; ++i) {
    bn_fold_kernel<<<(cout_[i] + 255) / 256, 256, 0, stream>>>(
        L(bases[i], 1), L(bases[i], 2), L(bases[i], 3), L(bases[i], 4),
        L(bases[i], 5), sc[i], sh[i], cout_[i]);
    int nthr = (cout_[i] / 16) * nkk_[i] * 32;
    wfrag_kernel<<<(nthr + 255) / 256, 256, 0, stream>>>(
        L(bases[i], 0), cout_[i], cin_[i], nkk_[i], A[i]);
  }

  init_out_kernel<<<(out_size + 255) / 256, 256, 0, stream>>>((float*)d_out, out_size);

  fps_kernel<8192, 256><<<kB, 256, 0, stream>>>(x, c1, kG1);
  fps_kernel<512, 256><<<kB, 256, 0, stream>>>(c1, c2, kG2);
  knn_group_kernel<<<kB * kG2, 64, 0, stream>>>(c1, c2, nb2h);
  mlp2_kernel<<<kB * kG2, 256, 0, stream>>>(nb2h,
      A[0], A[1], A[2], sc[0], sh[0], sc[1], sh[1], sc[2], sh[2], act3);
  mlp3_kernel<<<kB * 8, 256, 0, stream>>>(act3,
      A[3], A[4], A[5], sc[3], sh[3], sc[4], sh[4], sc[5], sh[5], (float*)d_out);
}
